// VDEncoder_26414048871114
// MI455X (gfx1250) — matrix operand order repaired and checked by reading, not compiled, not run on hardware
//
#include <hip/hip_runtime.h>
#include <hip/hip_bf16.h>

typedef _Float16 v16h __attribute__((ext_vector_type(16)));
typedef _Float16 v8h  __attribute__((ext_vector_type(8)));
typedef float    v8f  __attribute__((ext_vector_type(8)));

#define B_DIM 64
#define T_DIM 2048
#define INF_DIM 128
#define H_DIM 64
#define G4 256   // 4*H

// -------------------------------------------------------------------------
// Out[M,N] = A[M,K] @ W[N,K]^T + bias0[N] + bias1[N]
// One wave per 16x16 output tile, f16 WMMA, fp32 accumulate.
// A-operand layout (16-bit, 16x32): lane m=l15; VGPR0..3 = K hi*16+0..7,
// VGPR4..7 = K hi*16+8..15.  B-operand (32x16): lane n=l15, 16
// contiguous K values starting at hi*16 (row-major W[N,K] => contiguous).
// -------------------------------------------------------------------------
__global__ __launch_bounds__(256) void gemm_bias_f16wmma(
    const float* __restrict__ A, const float* __restrict__ W,
    const float* __restrict__ bias0, const float* __restrict__ bias1,
    float* __restrict__ Out, int M, int N, int K)
{
  const int lane = threadIdx.x & 31;
  const int wid  = threadIdx.x >> 5;
  const int tile = blockIdx.x * 8 + wid;
  const int ntn  = N >> 4;
  const int mt   = tile / ntn;
  const int nt   = tile - mt * ntn;
  if (mt * 16 >= M) return;
  const int l15 = lane & 15;
  const int hi  = lane >> 4;

  const float* arow = A + (size_t)(mt * 16 + l15) * K;
  const float* wrow = W + (size_t)(nt * 16 + l15) * K;

  v8f acc = {};
  for (int k0 = 0; k0 < K; k0 += 32) {
    v16h a, b;
    const int ka = k0 + hi * 16;
    #pragma unroll
    for (int p = 0; p < 8; ++p) {
      a[p]     = (_Float16)arow[ka + p];
      a[p + 8] = (_Float16)arow[ka + 8 + p];
    }
    const int kb = k0 + hi * 16;
    #pragma unroll
    for (int p = 0; p < 16; ++p)
      b[p] = (_Float16)wrow[kb + p];
    acc = __builtin_amdgcn_wmma_f32_16x16x32_f16(false, a, false, b,
                                                 (short)0, acc, false, false);
  }
  const int   n  = nt * 16 + l15;
  const float bs = bias0[n] + bias1[n];
  const int   m0 = mt * 16 + hi * 8;   // C layout: lane holds rows m0..m0+7, col n
  #pragma unroll
  for (int r = 0; r < 8; ++r)
    Out[(size_t)(m0 + r) * N + n] = acc[r] + bs;
}

// -------------------------------------------------------------------------
// Sequential LSTM scan. grid = 4 workgroups (16 batch rows each),
// block = 128 threads (4 waves). Wave jg computes ALL FOUR gates for hidden
// units jg*16..jg*16+15, so the c/h update is lane-local. W_hh^T B-tiles are
// register-resident (f16) for the whole scan; h lives in 2.3KB of LDS.
// -------------------------------------------------------------------------
__global__ __launch_bounds__(128) void lstm_scan(
    const float* __restrict__ xg,    // [B, T, 4H] pre-activations (x@W_ih^T + biases)
    const float* __restrict__ W_hh,  // [4H, H]
    const float* __restrict__ mask,  // [B, H]  (applied to output only)
    float* __restrict__ out,         // [B, T, H]
    int T, int apply_relu)
{
  __shared__ __attribute__((aligned(16))) _Float16 hlds[16][72]; // pad->no bank conflicts
  const int wg   = blockIdx.x;        // rows wg*16 .. wg*16+15
  const int jg   = threadIdx.x >> 5;  // wave: hidden-unit group
  const int lane = threadIdx.x & 31;
  const int l15  = lane & 15;
  const int hi   = lane >> 4;
  const int j    = jg * 16 + l15;     // this lane's hidden unit (B/C column)

  // Preload W_hh^T tiles: gate q occupies columns q*64 + j. Two K=32 halves.
  v16h breg[4][2];
  #pragma unroll
  for (int q = 0; q < 4; ++q) {
    const int n = q * 64 + j;
    #pragma unroll
    for (int kh = 0; kh < 2; ++kh)
      #pragma unroll
      for (int e = 0; e < 16; ++e)
        breg[q][kh][e] = (_Float16)W_hh[(size_t)n * H_DIM + (kh * 32 + hi * 16 + e)];
  }

  // Lane-persistent state: rows b = wg*16 + hi*8 + r (r=0..7), column j.
  float c_reg[8], m_reg[8];
  #pragma unroll
  for (int r = 0; r < 8; ++r) {
    c_reg[r] = 0.f;
    m_reg[r] = mask[(size_t)(wg * 16 + hi * 8 + r) * H_DIM + j];
  }

  for (int idx = threadIdx.x; idx < 16 * 64; idx += 128)
    hlds[idx >> 6][idx & 63] = (_Float16)0.f;
  __syncthreads();

  union V16 { v16h v; v8h h[2]; };

  for (int t = 0; t < T; ++t) {
    // xg gate pre-activations: independent of the recurrence -> issue first
    float xgr[4][8];
    #pragma unroll
    for (int r = 0; r < 8; ++r) {
      const size_t base = ((size_t)(wg * 16 + hi * 8 + r) * T + t) * G4;
      #pragma unroll
      for (int q = 0; q < 4; ++q)
        xgr[q][r] = xg[base + q * 64 + j];
    }

    // A-operand (h, 16x64 f16) from LDS in contiguous K order (k = 16*hi + s), same as breg
    V16 a0, a1;
    const int ks = hi * 16;
    a0.h[0] = *(const v8h*)&hlds[l15][ks];
    a0.h[1] = *(const v8h*)&hlds[l15][8 + ks];
    a1.h[0] = *(const v8h*)&hlds[l15][32 + ks];
    a1.h[1] = *(const v8h*)&hlds[l15][40 + ks];

    v8f acc[4];
    #pragma unroll
    for (int q = 0; q < 4; ++q) {
      v8f z = {};
      z = __builtin_amdgcn_wmma_f32_16x16x32_f16(false, a0.v, false, breg[q][0],
                                                 (short)0, z, false, false);
      z = __builtin_amdgcn_wmma_f32_16x16x32_f16(false, a1.v, false, breg[q][1],
                                                 (short)0, z, false, false);
      acc[q] = z;
    }

    __syncthreads();   // all h reads of this step complete before updates

    #pragma unroll
    for (int r = 0; r < 8; ++r) {
      const float gi = acc[0][r] + xgr[0][r];
      const float gf = acc[1][r] + xgr[1][r];
      const float gg = acc[2][r] + xgr[2][r];
      const float go = acc[3][r] + xgr[3][r];
      const float si = 1.f / (1.f + __expf(-gi));
      const float sf = 1.f / (1.f + __expf(-gf));
      const float so = 1.f / (1.f + __expf(-go));
      const float c  = sf * c_reg[r] + si * tanhf(gg);
      c_reg[r] = c;
      const float h = so * tanhf(c);
      hlds[hi * 8 + r][j] = (_Float16)h;           // recurrence uses UNmasked h
      float ov = h * m_reg[r];                     // variational dropout on output
      if (apply_relu) ov = fmaxf(ov, 0.f);
      out[((size_t)(wg * 16 + hi * 8 + r) * T + t) * H_DIM + j] = ov;
    }
    __syncthreads();   // h updates visible before next step's reads
  }
}

extern "C" void kernel_launch(void* const* d_in, const int* in_sizes, int n_in,
                              void* d_out, int out_size, void* d_ws, size_t ws_size,
                              hipStream_t stream) {
  const float* x     = (const float*)d_in[0];
  const float* W_ih1 = (const float*)d_in[1];
  const float* W_hh1 = (const float*)d_in[2];
  const float* b_ih1 = (const float*)d_in[3];
  const float* b_hh1 = (const float*)d_in[4];
  const float* W_ih2 = (const float*)d_in[5];
  const float* W_hh2 = (const float*)d_in[6];
  const float* b_ih2 = (const float*)d_in[7];
  const float* b_hh2 = (const float*)d_in[8];
  const float* mask1 = (const float*)d_in[9];
  const float* mask2 = (const float*)d_in[10];
  float* out = (float*)d_out;

  float* xg  = (float*)d_ws;                               // [B*T, 256] (reused)
  float* h1m = xg + (size_t)B_DIM * T_DIM * G4;            // [B*T, 64]

  const int M     = B_DIM * T_DIM;                         // 131072
  const int tiles = (M / 16) * (G4 / 16);                  // 131072 tiles
  const int grid  = tiles / 8;                             // 8 waves (tiles) / block

  // Layer 1: input projection, scan, masked output
  gemm_bias_f16wmma<<<grid, 256, 0, stream>>>(x, W_ih1, b_ih1, b_hh1, xg, M, G4, INF_DIM);
  lstm_scan<<<4, 128, 0, stream>>>(xg, W_hh1, mask1, h1m, T_DIM, 0);

  // Layer 2: input projection (K=64), scan, mask + ReLU into d_out
  gemm_bias_f16wmma<<<grid, 256, 0, stream>>>(h1m, W_ih2, b_ih2, b_hh2, xg, M, G4, H_DIM);
  lstm_scan<<<4, 128, 0, stream>>>(xg, W_hh2, mask2, out, T_DIM, 1);
}
